// Transformer_25829933318292
// MI455X (gfx1250) — matrix operand order repaired and checked by reading, not compiled, not run on hardware
//
#include <hip/hip_runtime.h>

// ---------------------------------------------------------------------------
// CDNA5 (gfx1250) transformer forward. All heavy GEMMs + attention on
// v_wmma_f32_16x16x32_bf16 (f32 accumulate). wave32 throughout.
// A-tiles staged with global_load_async_to_lds_b128 (ASYNCcnt path).
// ---------------------------------------------------------------------------

typedef __attribute__((ext_vector_type(16))) __bf16 v16bf;
typedef __attribute__((ext_vector_type(8)))  float  v8f;
typedef __attribute__((ext_vector_type(4)))  unsigned int u32x4;
typedef __attribute__((ext_vector_type(4)))  float  f32x4;

union Frag16 { u32x4 q[2]; v16bf v; };   // 16 bf16 = 32 bytes = one WMMA A/B operand
union PkBf2  { __bf16 h[2]; unsigned u; };

#define B_   8
#define T_   1024
#define D_   512
#define H_   8
#define DH_  64
#define L_   6
#define OBS_ 64
#define IN_  66
#define NTOK (B_ * T_)

__device__ __forceinline__ void async_wait0() {
#if __has_builtin(__builtin_amdgcn_s_wait_asynccnt)
  __builtin_amdgcn_s_wait_asynccnt(0);
#else
  asm volatile("s_wait_asynccnt 0x0" ::: "memory");
#endif
}

// ---------------------------------------------------------------------------
// Embedding: x[tok,:] = [inp(65), loc] @ embed_W  (tiny: 66-deep K, scalar FMA)
// ---------------------------------------------------------------------------
__global__ __launch_bounds__(128) void embed_kernel(const float* __restrict__ inp,
                                                    const float* __restrict__ W,
                                                    float* __restrict__ x) {
  int tok = blockIdx.x;
  int t   = tok & (T_ - 1);
  float loc = -1.0f + 2.0f * (float)t / (float)(T_ - 1);
  int d0 = threadIdx.x;                       // 128 threads x 4 columns
  float acc[4] = {0.f, 0.f, 0.f, 0.f};
  const float* row = inp + (size_t)tok * (OBS_ + 1);
  for (int kk = 0; kk < IN_; kk++) {
    float f = (kk < OBS_ + 1) ? row[kk] : loc;
    const float* wr = W + kk * D_ + d0;
#pragma unroll
    for (int j = 0; j < 4; j++) acc[j] += f * wr[j * 128];
  }
#pragma unroll
  for (int j = 0; j < 4; j++) x[(size_t)tok * D_ + d0 + j * 128] = acc[j];
}

// ---------------------------------------------------------------------------
// LayerNorm fp32 -> bf16 (one wave per token, 16 elems/lane, shfl reductions)
// ---------------------------------------------------------------------------
__global__ __launch_bounds__(256) void layernorm_bf16_kernel(const float* __restrict__ x,
                                                             const float* __restrict__ g,
                                                             const float* __restrict__ b,
                                                             __bf16* __restrict__ out) {
  int wave = threadIdx.x >> 5, lane = threadIdx.x & 31;
  int tok = blockIdx.x * 8 + wave;
  const float* xr = x + (size_t)tok * D_;
  float vals[16], s = 0.f, s2 = 0.f;
#pragma unroll
  for (int i = 0; i < 16; i++) { float f = xr[i * 32 + lane]; vals[i] = f; s += f; s2 += f * f; }
#pragma unroll
  for (int off = 1; off < 32; off <<= 1) { s += __shfl_xor(s, off, 32); s2 += __shfl_xor(s2, off, 32); }
  float mean = s * (1.0f / D_);
  float var  = s2 * (1.0f / D_) - mean * mean;
  float rstd = rsqrtf(var + 1e-5f);
  __bf16* orow = out + (size_t)tok * D_;
#pragma unroll
  for (int i = 0; i < 16; i++) {
    int d = i * 32 + lane;
    orow[d] = (__bf16)((vals[i] - mean) * rstd * g[d] + b[d]);
  }
}

// ---------------------------------------------------------------------------
// GEMM: out[M,N] = A_bf16[M,K] @ W_f32[K,N] + bias, fused epilogues.
// 4 waves / block, 64x64 block tile, 32x32 per wave (2x2 WMMA C tiles), KB=32.
// A tile: global_load_async_to_lds_b128 (direct global->LDS DMA, ASYNCcnt).
// W tile: fp32->bf16 convert + transpose during staging, packed b32 DS stores,
// so B-frag reads are pairs of k-contiguous 16B LDS vectors matching the 32x16 B layout.
// ---------------------------------------------------------------------------
enum { EPI_BF16 = 0, EPI_RES_F32 = 1, EPI_GELU_BF16 = 2, EPI_F32 = 3 };

__global__ __launch_bounds__(128) void gemm_bf16_wmma_kernel(
    const __bf16* __restrict__ A, const float* __restrict__ W,
    const float* __restrict__ bias, const float* __restrict__ resid,
    void* __restrict__ outp, int M, int N, int K, int epi) {
  __shared__ __bf16 As[64][32];
  __shared__ __bf16 Bt[64][32];               // transposed: [n][k]
  int tid = threadIdx.x;
  int wave = tid >> 5, lane = tid & 31, lm = lane & 15, half = lane >> 4;
  int wr = wave >> 1, wc = wave & 1;
  int bm = blockIdx.y * 64, bn = blockIdx.x * 64;

  v8f c[2][2] = {};

  int arow = tid >> 1, aseg = tid & 1;        // A staging: 64 rows x (2 x 16 bf16)
  int bkp = tid & 15, bng = tid >> 4;         // B staging: k-pair 0..15, n-octet 0..7

  for (int kb = 0; kb < K; kb += 32) {
    // ---- stage A tile: async DMA global -> LDS (2 x b128 per thread) ----
    {
      unsigned lds_dst = (unsigned)(unsigned long long)(void*)&As[arow][aseg * 16];
      unsigned long long gsrc =
          (unsigned long long)(A + (size_t)(bm + arow) * K + kb + aseg * 16);
      asm volatile("global_load_async_to_lds_b128 %0, %1, off"
                   :: "v"(lds_dst), "v"(gsrc) : "memory");
      asm volatile("global_load_async_to_lds_b128 %0, %1, off offset:16"
                   :: "v"(lds_dst), "v"(gsrc) : "memory");
    }
    // ---- stage B tile (fp32 -> bf16, transpose, packed pair stores) ----
    {
      const float* w0 = W + (size_t)(kb + 2 * bkp) * N + bn + bng * 8;
      const float* w1 = w0 + N;
      if (kb + 32 < K) __builtin_prefetch(w0 + 32 * N, 0, 1);  // next K tile of W
      f32x4 r0a = *(const f32x4*)w0, r0b = *(const f32x4*)(w0 + 4);
      f32x4 r1a = *(const f32x4*)w1, r1b = *(const f32x4*)(w1 + 4);
#pragma unroll
      for (int i = 0; i < 8; i++) {
        float lo = (i < 4) ? r0a[i] : r0b[i - 4];
        float hi = (i < 4) ? r1a[i] : r1b[i - 4];
        PkBf2 p; p.h[0] = (__bf16)lo; p.h[1] = (__bf16)hi;
        *(unsigned*)&Bt[bng * 8 + i][2 * bkp] = p.u;
      }
    }
    async_wait0();                            // A tile landed in LDS (this wave)
    __syncthreads();

    Frag16 a[2], bf[2];
#pragma unroll
    for (int i = 0; i < 2; i++) {             // A 16x32: e0..7 -> K=8h.., e8..15 -> K=16+8h..
      int m = wr * 32 + i * 16 + lm;
      a[i].q[0] = *(const u32x4*)&As[m][8 * half];
      a[i].q[1] = *(const u32x4*)&As[m][16 + 8 * half];
    }
#pragma unroll
    for (int j = 0; j < 2; j++) {             // B 32x16: lane=n, e0..7 -> K=8h.., e8..15 -> K=16+8h..
      int n = wc * 32 + j * 16 + lm;
      bf[j].q[0] = *(const u32x4*)&Bt[n][8 * half];
      bf[j].q[1] = *(const u32x4*)&Bt[n][16 + 8 * half];
    }
#pragma unroll
    for (int i = 0; i < 2; i++)
#pragma unroll
      for (int j = 0; j < 2; j++)
        c[i][j] = __builtin_amdgcn_wmma_f32_16x16x32_bf16(
            false, a[i].v, false, bf[j].v, (short)0, c[i][j], false, false);
    __syncthreads();
  }

  // ---- epilogue (C layout: row = r + 8*half, col = lane&15) ----
#pragma unroll
  for (int i = 0; i < 2; i++)
#pragma unroll
    for (int j = 0; j < 2; j++)
#pragma unroll
      for (int r = 0; r < 8; r++) {
        int row = bm + wr * 32 + i * 16 + r + 8 * half;
        int col = bn + wc * 32 + j * 16 + lm;
        float val = c[i][j][r] + bias[col];
        size_t idx = (size_t)row * N + col;
        if (epi == EPI_BF16)          ((__bf16*)outp)[idx] = (__bf16)val;
        else if (epi == EPI_RES_F32)  ((float*)outp)[idx]  = resid[idx] + val;
        else if (epi == EPI_GELU_BF16)((__bf16*)outp)[idx] =
            (__bf16)(0.5f * val * (1.0f + erff(val * 0.70710678f)));
        else                          ((float*)outp)[idx]  = val;
      }
}

// ---------------------------------------------------------------------------
// Flash attention: one wave per (b, h, 16-query tile). Per 32-key chunk:
// 4 WMMA for S = Q K^T, online softmax via width-16 shfl (C layout keeps each
// row inside one half-wave), P bounced through 1KB LDS to A layout,
// 4 WMMA for O += P V (V staged transposed in LDS). Causal mask fused.
// ---------------------------------------------------------------------------
__global__ __launch_bounds__(32) void attn_flash_wmma_kernel(
    const __bf16* __restrict__ q, const __bf16* __restrict__ k,
    const __bf16* __restrict__ v, __bf16* __restrict__ y) {
  __shared__ __bf16 Plds[16][32];
  __shared__ __bf16 Vlds[64][32];             // [feature][key-in-chunk]
  int lane = threadIdx.x, lm = lane & 15, half = lane >> 4;
  int qb = blockIdx.x * 16, h = blockIdx.y, b = blockIdx.z;
  int bT = b * T_;
  size_t hoff = (size_t)h * DH_;

  Frag16 qa[2];
  {
    const __bf16* qp = q + (size_t)(bT + qb + lm) * D_ + hoff;
#pragma unroll
    for (int cc = 0; cc < 2; cc++) {
      qa[cc].q[0] = *(const u32x4*)(qp + 32 * cc + 8 * half);
      qa[cc].q[1] = *(const u32x4*)(qp + 32 * cc + 16 + 8 * half);
    }
  }
  v8f o[4] = {};
  float mrow[8], lrow[8];
#pragma unroll
  for (int r = 0; r < 8; r++) { mrow[r] = -1e30f; lrow[r] = 0.f; }

  int nch = (qb + 16 + 31) >> 5;
  for (int ch = 0; ch < nch; ch++) {
    int kb = ch * 32;
    // stage V^T: lane owns one key row (64 bf16 = 8x128b loads), scatter to LDS
    {
      union { u32x4 qv[8]; __bf16 e[64]; } vr;
      const __bf16* vp = v + (size_t)(bT + kb + lane) * D_ + hoff;
#pragma unroll
      for (int i = 0; i < 8; i++) vr.qv[i] = *(const u32x4*)(vp + i * 8);
#pragma unroll
      for (int f = 0; f < 64; f++) Vlds[f][lane] = vr.e[f];
    }
    // S = Q K^T : two 16x16 score tiles, K-dim 64 as 2x32
    v8f s[2];
#pragma unroll
    for (int j = 0; j < 2; j++) {
      Frag16 kf0, kf1;
      const __bf16* kp = k + (size_t)(bT + kb + 16 * j + lm) * D_ + hoff;
      kf0.q[0] = *(const u32x4*)(kp + 8 * half);
      kf0.q[1] = *(const u32x4*)(kp + 16 + 8 * half);
      kf1.q[0] = *(const u32x4*)(kp + 32 + 8 * half);
      kf1.q[1] = *(const u32x4*)(kp + 32 + 16 + 8 * half);
      v8f z = {};
      z = __builtin_amdgcn_wmma_f32_16x16x32_bf16(false, qa[0].v, false, kf0.v, (short)0, z, false, false);
      z = __builtin_amdgcn_wmma_f32_16x16x32_bf16(false, qa[1].v, false, kf1.v, (short)0, z, false, false);
      s[j] = z;
    }
    // scale + causal mask + online softmax
#pragma unroll
    for (int r = 0; r < 8; r++) {
      int row = qb + r + 8 * half;
      float s0 = s[0][r] * 0.125f, s1 = s[1][r] * 0.125f;
      if (kb + lm > row)      s0 = -1e30f;
      if (kb + 16 + lm > row) s1 = -1e30f;
      float mx = fmaxf(s0, s1);
#pragma unroll
      for (int off = 1; off < 16; off <<= 1) mx = fmaxf(mx, __shfl_xor(mx, off, 16));
      float mn = fmaxf(mrow[r], mx);
      float fr = __expf(mrow[r] - mn);
      mrow[r] = mn;
      float p0 = __expf(s0 - mn), p1 = __expf(s1 - mn);
      float ps = p0 + p1;
#pragma unroll
      for (int off = 1; off < 16; off <<= 1) ps += __shfl_xor(ps, off, 16);
      lrow[r] = lrow[r] * fr + ps;
#pragma unroll
      for (int n = 0; n < 4; n++) o[n][r] *= fr;
      Plds[r + 8 * half][lm]      = (__bf16)p0;   // C layout -> LDS
      Plds[r + 8 * half][16 + lm] = (__bf16)p1;
    }
    // P back as A fragment (16 queries x 32 keys); same-wave DS ops are in order
    Frag16 pa;
    pa.q[0] = *(const u32x4*)&Plds[lm][8 * half];
    pa.q[1] = *(const u32x4*)&Plds[lm][16 + 8 * half];
    // O += P V
#pragma unroll
    for (int n = 0; n < 4; n++) {
      Frag16 vb;
      vb.q[0] = *(const u32x4*)&Vlds[16 * n + lm][8 * half];
      vb.q[1] = *(const u32x4*)&Vlds[16 * n + lm][16 + 8 * half];
      o[n] = __builtin_amdgcn_wmma_f32_16x16x32_bf16(false, pa.v, false, vb.v, (short)0, o[n], false, false);
    }
  }
  // normalize + write bf16
#pragma unroll
  for (int r = 0; r < 8; r++) {
    float inv = 1.0f / lrow[r];
    int row = qb + r + 8 * half;
    __bf16* yp = y + (size_t)(bT + row) * D_ + hoff;
#pragma unroll
    for (int n = 0; n < 4; n++) yp[16 * n + lm] = (__bf16)(o[n][r] * inv);
  }
}

// ---------------------------------------------------------------------------
// Split head output into mu / softplus(log_std)+MIN_STD
// ---------------------------------------------------------------------------
__global__ __launch_bounds__(256) void head_split_kernel(const float* __restrict__ ho,
                                                         float* __restrict__ out) {
  int idx = blockIdx.x * 256 + threadIdx.x;   // 0 .. NTOK*64-1
  int t = idx >> 6, j = idx & 63;
  float mu = ho[(size_t)t * 128 + j];
  float ls = ho[(size_t)t * 128 + 64 + j];
  float sp = (ls > 20.f) ? ls : log1pf(__expf(ls));
  out[idx] = mu;
  out[(size_t)NTOK * 64 + idx] = sp + 0.01f;
}

// ---------------------------------------------------------------------------
extern "C" void kernel_launch(void* const* d_in, const int* in_sizes, int n_in,
                              void* d_out, int out_size, void* d_ws, size_t ws_size,
                              hipStream_t stream) {
  const float* inp    = (const float*)d_in[0];
  const float* embedW = (const float*)d_in[1];
  const float* Wq = (const float*)d_in[2];
  const float* Wk = (const float*)d_in[3];
  const float* Wv = (const float*)d_in[4];
  const float* Wp = (const float*)d_in[5];
  const float* bq = (const float*)d_in[6];
  const float* bk = (const float*)d_in[7];
  const float* bv = (const float*)d_in[8];
  const float* bp = (const float*)d_in[9];
  const float* ln1g = (const float*)d_in[10];
  const float* ln1b = (const float*)d_in[11];
  const float* ln2g = (const float*)d_in[12];
  const float* ln2b = (const float*)d_in[13];
  const float* W1 = (const float*)d_in[14];
  const float* b1 = (const float*)d_in[15];
  const float* W2 = (const float*)d_in[16];
  const float* b2 = (const float*)d_in[17];
  const float* lnfg = (const float*)d_in[18];
  const float* lnfb = (const float*)d_in[19];
  const float* headW = (const float*)d_in[20];
  const float* headb = (const float*)d_in[21];
  (void)in_sizes; (void)n_in; (void)out_size; (void)ws_size;

  char* wsp = (char*)d_ws;
  auto carve = [&](size_t bytes) -> char* {
    char* p = wsp; wsp += (bytes + 255) & ~(size_t)255; return p;
  };
  float*  x      = (float*)carve((size_t)NTOK * D_ * 4);
  __bf16* h_bf   = (__bf16*)carve((size_t)NTOK * D_ * 2);
  __bf16* q_bf   = (__bf16*)carve((size_t)NTOK * D_ * 2);
  __bf16* k_bf   = (__bf16*)carve((size_t)NTOK * D_ * 2);
  __bf16* v_bf   = (__bf16*)carve((size_t)NTOK * D_ * 2);
  __bf16* y_bf   = (__bf16*)carve((size_t)NTOK * D_ * 2);
  __bf16* mid_bf = (__bf16*)carve((size_t)NTOK * 4 * D_ * 2);
  float*  head_o = (float*)carve((size_t)NTOK * 128 * 4);

  auto gemm = [&](const __bf16* A, const float* Wm, const float* bias, const float* resid,
                  void* outp, int M, int N, int K, int epi) {
    dim3 grid(N / 64, M / 64);
    gemm_bf16_wmma_kernel<<<grid, 128, 0, stream>>>(A, Wm, bias, resid, outp, M, N, K, epi);
  };

  embed_kernel<<<NTOK, 128, 0, stream>>>(inp, embedW, x);

  for (int l = 0; l < L_; l++) {
    size_t wo = (size_t)l * D_ * D_;
    layernorm_bf16_kernel<<<NTOK / 8, 256, 0, stream>>>(x, ln1g + l * D_, ln1b + l * D_, h_bf);
    gemm(h_bf, Wq + wo, bq + l * D_, nullptr, q_bf, NTOK, D_, D_, EPI_BF16);
    gemm(h_bf, Wk + wo, bk + l * D_, nullptr, k_bf, NTOK, D_, D_, EPI_BF16);
    gemm(h_bf, Wv + wo, bv + l * D_, nullptr, v_bf, NTOK, D_, D_, EPI_BF16);
    attn_flash_wmma_kernel<<<dim3(T_ / 16, H_, B_), 32, 0, stream>>>(q_bf, k_bf, v_bf, y_bf);
    gemm(y_bf, Wp + wo, bp + l * D_, x, x, NTOK, D_, D_, EPI_RES_F32);
    layernorm_bf16_kernel<<<NTOK / 8, 256, 0, stream>>>(x, ln2g + l * D_, ln2b + l * D_, h_bf);
    gemm(h_bf, W1 + (size_t)l * D_ * 4 * D_, b1 + (size_t)l * 4 * D_, nullptr,
         mid_bf, NTOK, 4 * D_, D_, EPI_GELU_BF16);
    gemm(mid_bf, W2 + (size_t)l * 4 * D_ * D_, b2 + l * D_, x, x, NTOK, D_, 4 * D_, EPI_RES_F32);
  }
  layernorm_bf16_kernel<<<NTOK / 8, 256, 0, stream>>>(x, lnfg, lnfb, h_bf);
  gemm(h_bf, headW, headb, nullptr, head_o, NTOK, 128, D_, EPI_F32);
  head_split_kernel<<<(NTOK * 64) / 256, 256, 0, stream>>>(head_o, (float*)d_out);
}
